// Attention_2430951489678
// MI455X (gfx1250) — matrix operand order repaired and checked by reading, not compiled, not run on hardware
//
#include <hip/hip_runtime.h>
#include <hip/hip_bf16.h>
#include <cstdint>

// ---------------------------------------------------------------------------
// Types for CDNA5 WMMA (wave32): V_WMMA_F32_16X16X32_BF16
// ---------------------------------------------------------------------------
typedef __attribute__((ext_vector_type(16))) __bf16 v16bf;
typedef __attribute__((ext_vector_type(8)))  float  v8f;

struct Frag {
    union {
        v16bf    bf;
        uint32_t d[8];
    };
};

__device__ __forceinline__ uint16_t f2bf(float f) {
    uint32_t u = __float_as_uint(f);
    uint32_t r = u + 0x7FFFu + ((u >> 16) & 1u);   // round-to-nearest-even
    return (uint16_t)(r >> 16);
}

__device__ __forceinline__ v8f wmma_bf16(const Frag& a, const Frag& b, v8f c) {
    return __builtin_amdgcn_wmma_f32_16x16x32_bf16(
        /*neg_a=*/false, a.bf, /*neg_b=*/false, b.bf,
        /*c_mod=*/(short)0, c, /*reuse_a=*/false, /*reuse_b=*/false);
}

// A-matrix fragment (16x32 bf16, MxK), source row-major [M][ld], base at (m0,k0).
// ISA layout: lanes 0-15 -> M=lane, K {0..7,16..23}; lanes 16-31 -> M=lane-16, K {8..15,24..31}
__device__ __forceinline__ Frag loadA(const uint16_t* base, int ld, int lane) {
    int half = lane >> 4, mr = lane & 15;
    const uint16_t* p = base + (size_t)mr * ld + (half ? 8 : 0);
    Frag f;
    *(uint4*)&f.d[0] = *(const uint4*)p;          // K +0..7   (or +8..15)
    *(uint4*)&f.d[4] = *(const uint4*)(p + 16);   // K +16..23 (or +24..31)
    return f;
}

// B-matrix fragment (32x16 bf16, KxN) from B^T storage row-major [N][ld], base at (n0,k0).
// ISA layout: lanes 0-15 -> N=lane, K {0..7,16..23}; lanes 16-31 -> N=lane-16, K {8..15,24..31}
__device__ __forceinline__ Frag loadBt(const uint16_t* base, int ld, int lane) {
    int half = lane >> 4, nr = lane & 15;
    const uint16_t* p = base + (size_t)nr * ld + (half ? 8 : 0);
    Frag f;
    *(uint4*)&f.d[0] = *(const uint4*)p;
    *(uint4*)&f.d[4] = *(const uint4*)(p + 16);
    return f;
}

// One 16-byte async Global->LDS copy (CDNA5: GLOBAL_LOAD_ASYNC_TO_LDS_B128,
// tracked by ASYNCcnt). dst is the LDS byte address (low 32 bits of the
// generic pointer == LDS offset per the aperture mapping), src is a 64-bit
// global address in a VGPR pair; GV addressing mode (SADDR = off).
__device__ __forceinline__ void async_copy16(uint32_t lds_addr, const void* gsrc) {
    asm volatile("global_load_async_to_lds_b128 %0, %1, off"
                 :: "v"(lds_addr), "v"((uint64_t)(uintptr_t)gsrc)
                 : "memory");
}

// ---------------------------------------------------------------------------
// Constants
// ---------------------------------------------------------------------------
static constexpr int B_   = 2;
static constexpr int N_   = 4096;
static constexpr int D_   = 512;
static constexpr int H_   = 8;
static constexpr int HD_  = 64;
static constexpr int M_   = B_ * N_;      // 8192 rows
static constexpr int QKVN = 3 * D_;       // 1536

// ---------------------------------------------------------------------------
// fp32 -> bf16 conversion
// ---------------------------------------------------------------------------
__global__ void cvt_f32_bf16(const float* __restrict__ src,
                             uint16_t* __restrict__ dst, int n) {
    int i = blockIdx.x * blockDim.x + threadIdx.x;
    if (i < n) dst[i] = f2bf(src[i]);
}

// ---------------------------------------------------------------------------
// QKV GEMM: [8192,512]bf16 x [1536,512]^T -> bf16 Q[bh][n][64], K[bh][n][64],
// Vt[bh][64][n]. Wave tile 32x64 (2 A-frags, 8 accumulators, B reused x2).
// ---------------------------------------------------------------------------
__global__ __launch_bounds__(256)
void qkv_gemm(const uint16_t* __restrict__ xb, const uint16_t* __restrict__ wb,
              uint16_t* __restrict__ Qb, uint16_t* __restrict__ Kb,
              uint16_t* __restrict__ Vt) {
    const int lane = threadIdx.x & 31, warp = threadIdx.x >> 5;
    const int gw = blockIdx.x * 8 + warp;
    const int NT = QKVN / 64;                       // 24 column tiles
    const int mt = gw / NT, nt = gw % NT;
    const int m0 = mt * 32, n0 = nt * 64;

    v8f acc0[4] = {{}, {}, {}, {}};
    v8f acc1[4] = {{}, {}, {}, {}};
    for (int k0 = 0; k0 < D_; k0 += 32) {
        Frag a0 = loadA(xb + (size_t)m0 * D_ + k0,        D_, lane);
        Frag a1 = loadA(xb + (size_t)(m0 + 16) * D_ + k0, D_, lane);
#pragma unroll
        for (int t = 0; t < 4; ++t) {
            Frag b = loadBt(wb + (size_t)(n0 + t * 16) * D_ + k0, D_, lane);
            acc0[t] = wmma_bf16(a0, b, acc0[t]);
            acc1[t] = wmma_bf16(a1, b, acc1[t]);
        }
    }

    const int half = lane >> 4, lc = lane & 15;
#pragma unroll
    for (int mi = 0; mi < 2; ++mi) {
#pragma unroll
        for (int t = 0; t < 4; ++t) {
#pragma unroll
            for (int r = 0; r < 8; ++r) {
                int m = m0 + mi * 16 + r + 8 * half;    // row in [0,8192)
                int n = n0 + t * 16 + lc;               // col in [0,1536)
                int b  = m >> 12, nn = m & (N_ - 1);
                int which = n >> 9, hh = (n >> 6) & 7, hd = n & 63;
                uint16_t v = f2bf(mi ? acc1[t][r] : acc0[t][r]);
                size_t bh = (size_t)(b * H_ + hh);
                if (which == 0)      Qb[(bh * N_ + nn) * HD_ + hd] = v;
                else if (which == 1) Kb[(bh * N_ + nn) * HD_ + hd] = v;
                else                 Vt[(bh * HD_ + hd) * N_ + nn] = v;
            }
        }
    }
}

// ---------------------------------------------------------------------------
// Flash attention. One wave per (bh, 16-query tile); the 8 waves of a block
// cover 8 consecutive query tiles of the SAME (b,h), so K/V tiles are staged
// once per block into LDS with double-buffered GLOBAL_LOAD_ASYNC_TO_LDS_B128
// (2 async instructions per wave per batch -> s_wait_asynccnt 0x2 retires the
// in-flight buffer). All inner-loop fragment reads are ds_read_b128.
// ---------------------------------------------------------------------------
__global__ __launch_bounds__(256)
void flash_attn(const uint16_t* __restrict__ Qb, const uint16_t* __restrict__ Kb,
                const uint16_t* __restrict__ Vt, uint16_t* __restrict__ att) {
    __shared__ alignas(16) uint16_t Klds[2][32 * 64];   // 2 x 4KB: [key][hd]
    __shared__ alignas(16) uint16_t Vlds[2][64 * 32];   // 2 x 4KB: [hd][key]
    __shared__ alignas(16) uint16_t plds[8][16 * 32];   // per-wave P swizzle

    const int tid  = threadIdx.x;
    const int lane = tid & 31, warp = tid >> 5;
    const int gw = blockIdx.x * 8 + warp;               // 0..4095
    const int bh = gw >> 8;                             // / (4096/16)
    const int m0 = (gw & 255) * 16;
    const int half = lane >> 4, lc = lane & 15;

    const uint16_t* Q = Qb + (size_t)bh * N_ * HD_;
    const uint16_t* K = Kb + (size_t)bh * N_ * HD_;
    const uint16_t* V = Vt + (size_t)bh * HD_ * N_;

    // Per-thread async-copy addressing: K tile (32 keys x 64 hd) is 4KB
    // contiguous -> thread copies 16B chunk `tid`. V tile is 64 rows of 64B
    // from Vt (row stride N_) -> thread copies row tid>>2, 16B part tid&3.
    const int vrow = tid >> 2, vpart = tid & 3;
    auto issue_tiles = [&](int buf, int kk) {
        const uint8_t* ksrc = (const uint8_t*)(K + (size_t)kk * HD_) + tid * 16;
        async_copy16((uint32_t)(uintptr_t)&Klds[buf][tid * 8], ksrc);
        const uint8_t* vsrc =
            (const uint8_t*)(V + (size_t)vrow * N_ + kk) + vpart * 16;
        async_copy16((uint32_t)(uintptr_t)&Vlds[buf][vrow * 32 + vpart * 8], vsrc);
    };

    // Q fragments for the whole tile (Hd = 64 -> two K=32 fragments), kept live.
    Frag qa0 = loadA(Q + (size_t)m0 * HD_ + 0,  HD_, lane);
    Frag qa1 = loadA(Q + (size_t)m0 * HD_ + 32, HD_, lane);

    v8f o0 = {}, o1 = {}, o2 = {}, o3 = {};
    float mrow[8], lrow[8];
#pragma unroll
    for (int r = 0; r < 8; ++r) { mrow[r] = -1e30f; lrow[r] = 0.0f; }

    const float scale = 0.125f;                         // 64^-0.5
    const int NIT = N_ / 32;                            // 128

    issue_tiles(0, 0);                                  // prologue batch
    for (int it = 0; it < NIT; ++it) {
        const int buf = it & 1;
        if (it + 1 < NIT) {
            issue_tiles(buf ^ 1, (it + 1) * 32);
            // 2 outstanding from next batch; <=2 means current buffer landed
            asm volatile("s_wait_asynccnt 0x2" ::: "memory");
        } else {
            asm volatile("s_wait_asynccnt 0x0" ::: "memory");
        }
        __syncthreads();                                // whole tile visible

        const uint16_t* Kt = &Klds[buf][0];             // [32][64]
        const uint16_t* Vl = &Vlds[buf][0];             // [64][32]

        // ---- S = Q K^T for 32 keys (two 16x16 C tiles), from LDS ----
        Frag kb0a = loadBt(Kt + 0 * 64 + 0,   64, lane);
        Frag kb0b = loadBt(Kt + 0 * 64 + 32,  64, lane);
        Frag kb1a = loadBt(Kt + 16 * 64 + 0,  64, lane);
        Frag kb1b = loadBt(Kt + 16 * 64 + 32, 64, lane);
        v8f s0 = {}, s1 = {};
        s0 = wmma_bf16(qa0, kb0a, s0);
        s0 = wmma_bf16(qa1, kb0b, s0);
        s1 = wmma_bf16(qa0, kb1a, s1);
        s1 = wmma_bf16(qa1, kb1b, s1);

        // ---- online softmax update (per row; 16 lanes share a row) ----
        float fac[8];
#pragma unroll
        for (int r = 0; r < 8; ++r) {
            float a  = s0[r] * scale;
            float b2 = s1[r] * scale;
            float mx = fmaxf(a, b2);
            mx = fmaxf(mx, __shfl_xor(mx, 1, 32));
            mx = fmaxf(mx, __shfl_xor(mx, 2, 32));
            mx = fmaxf(mx, __shfl_xor(mx, 4, 32));
            mx = fmaxf(mx, __shfl_xor(mx, 8, 32));
            float mn = fmaxf(mrow[r], mx);
            fac[r] = __expf(mrow[r] - mn);
            float p0 = __expf(a - mn);
            float p1 = __expf(b2 - mn);
            s0[r] = p0; s1[r] = p1;
            float rs = p0 + p1;
            rs += __shfl_xor(rs, 1, 32);
            rs += __shfl_xor(rs, 2, 32);
            rs += __shfl_xor(rs, 4, 32);
            rs += __shfl_xor(rs, 8, 32);
            lrow[r] = lrow[r] * fac[r] + rs;
            mrow[r] = mn;
        }
#pragma unroll
        for (int r = 0; r < 8; ++r) {
            o0[r] *= fac[r]; o1[r] *= fac[r];
            o2[r] *= fac[r]; o3[r] *= fac[r];
        }

        // ---- C-layout -> A-layout swizzle of P via per-wave LDS tile ----
#pragma unroll
        for (int r = 0; r < 8; ++r) {
            int row = r + 8 * half;
            plds[warp][row * 32 + lc]      = f2bf(s0[r]);
            plds[warp][row * 32 + 16 + lc] = f2bf(s1[r]);
        }
        Frag pa = loadA(&plds[warp][0], 32, lane);

        // ---- O += P * V  (Vlds rows are Hd -> perfect B fragments) ----
        Frag vb0 = loadBt(Vl + (0 * 16) * 32, 32, lane);
        Frag vb1 = loadBt(Vl + (1 * 16) * 32, 32, lane);
        Frag vb2 = loadBt(Vl + (2 * 16) * 32, 32, lane);
        Frag vb3 = loadBt(Vl + (3 * 16) * 32, 32, lane);
        o0 = wmma_bf16(pa, vb0, o0);
        o1 = wmma_bf16(pa, vb1, o1);
        o2 = wmma_bf16(pa, vb2, o2);
        o3 = wmma_bf16(pa, vb3, o3);

        __syncthreads();   // all reads done before next batch overwrites buf^1
    }

    // ---- normalize and scatter to att[B*N][512] at columns h*64.. ----
#pragma unroll
    for (int r = 0; r < 8; ++r) {
        float inv = 1.0f / lrow[r];
        o0[r] *= inv; o1[r] *= inv; o2[r] *= inv; o3[r] *= inv;
    }
    const int b = bh >> 3, h = bh & 7;
#pragma unroll
    for (int r = 0; r < 8; ++r) {
        size_t row = (size_t)b * N_ + m0 + r + 8 * half;
        size_t base = row * D_ + h * HD_;
        att[base + 0 * 16 + lc] = f2bf(o0[r]);
        att[base + 1 * 16 + lc] = f2bf(o1[r]);
        att[base + 2 * 16 + lc] = f2bf(o2[r]);
        att[base + 3 * 16 + lc] = f2bf(o3[r]);
    }
}

// ---------------------------------------------------------------------------
// Output projection: att[8192,512]bf16 x wproj[512,512]^T + bias -> fp32 out
// Wave tile 32x64.
// ---------------------------------------------------------------------------
__global__ __launch_bounds__(256)
void proj_gemm(const uint16_t* __restrict__ ab, const uint16_t* __restrict__ wb,
               const float* __restrict__ bias, float* __restrict__ out) {
    const int lane = threadIdx.x & 31, warp = threadIdx.x >> 5;
    const int gw = blockIdx.x * 8 + warp;
    const int NT = D_ / 64;                             // 8 column tiles
    const int mt = gw / NT, nt = gw % NT;
    const int m0 = mt * 32, n0 = nt * 64;

    v8f acc0[4] = {{}, {}, {}, {}};
    v8f acc1[4] = {{}, {}, {}, {}};
    for (int k0 = 0; k0 < D_; k0 += 32) {
        Frag a0 = loadA(ab + (size_t)m0 * D_ + k0,        D_, lane);
        Frag a1 = loadA(ab + (size_t)(m0 + 16) * D_ + k0, D_, lane);
#pragma unroll
        for (int t = 0; t < 4; ++t) {
            Frag b = loadBt(wb + (size_t)(n0 + t * 16) * D_ + k0, D_, lane);
            acc0[t] = wmma_bf16(a0, b, acc0[t]);
            acc1[t] = wmma_bf16(a1, b, acc1[t]);
        }
    }

    const int half = lane >> 4, lc = lane & 15;
#pragma unroll
    for (int t = 0; t < 4; ++t) {
        float bv = bias[n0 + t * 16 + lc];
#pragma unroll
        for (int r = 0; r < 8; ++r) {
            size_t row0 = (size_t)m0 + r + 8 * half;
            out[row0 * D_ + n0 + t * 16 + lc]        = acc0[t][r] + bv;
            out[(row0 + 16) * D_ + n0 + t * 16 + lc] = acc1[t][r] + bv;
        }
    }
}

// ---------------------------------------------------------------------------
// Host launch
// ---------------------------------------------------------------------------
extern "C" void kernel_launch(void* const* d_in, const int* in_sizes, int n_in,
                              void* d_out, int out_size, void* d_ws, size_t ws_size,
                              hipStream_t stream) {
    const float* x      = (const float*)d_in[0];
    const float* w_qkv  = (const float*)d_in[1];
    const float* w_proj = (const float*)d_in[2];
    const float* b_proj = (const float*)d_in[3];
    float* out = (float*)d_out;

    uint8_t* ws = (uint8_t*)d_ws;
    size_t off = 0;
    auto alloc = [&](size_t bytes) {
        void* p = ws + off;
        off += (bytes + 255) & ~(size_t)255;
        return p;
    };
    uint16_t* xb    = (uint16_t*)alloc((size_t)M_ * D_ * 2);
    uint16_t* wqkvb = (uint16_t*)alloc((size_t)QKVN * D_ * 2);
    uint16_t* wprjb = (uint16_t*)alloc((size_t)D_ * D_ * 2);
    uint16_t* Qb    = (uint16_t*)alloc((size_t)B_ * H_ * N_ * HD_ * 2);
    uint16_t* Kb    = (uint16_t*)alloc((size_t)B_ * H_ * N_ * HD_ * 2);
    uint16_t* Vt    = (uint16_t*)alloc((size_t)B_ * H_ * N_ * HD_ * 2);
    uint16_t* att   = (uint16_t*)alloc((size_t)M_ * D_ * 2);

    // fp32 -> bf16 staging
    {
        int n0 = M_ * D_, n1 = QKVN * D_, n2 = D_ * D_;
        cvt_f32_bf16<<<(n0 + 255) / 256, 256, 0, stream>>>(x,      xb,    n0);
        cvt_f32_bf16<<<(n1 + 255) / 256, 256, 0, stream>>>(w_qkv,  wqkvb, n1);
        cvt_f32_bf16<<<(n2 + 255) / 256, 256, 0, stream>>>(w_proj, wprjb, n2);
    }

    // QKV projection: (8192/32) * (1536/64) = 6144 wave tiles / 8 waves
    qkv_gemm<<<(M_ / 32) * (QKVN / 64) / 8, 256, 0, stream>>>(xb, wqkvb, Qb, Kb, Vt);

    // Flash attention: 16 (b,h) * 256 row-tiles = 4096 waves / 8 per block
    flash_attn<<<(B_ * H_ * (N_ / 16)) / 8, 256, 0, stream>>>(Qb, Kb, Vt, att);

    // Output projection: (8192/32) * (512/64) = 2048 wave tiles / 8
    proj_gemm<<<(M_ / 32) * (D_ / 64) / 8, 256, 0, stream>>>(att, wprjb, b_proj, out);
}
